// AttnBlock_78056735637626
// MI455X (gfx1250) — matrix operand order repaired and checked by reading, not compiled, not run on hardware
//
#include <hip/hip_runtime.h>

#define CH   256
#define NP   1024
#define NB   32
#define NG   32
#define CPG  8

typedef __attribute__((ext_vector_type(16))) __bf16 v16bf;
typedef __attribute__((ext_vector_type(8)))  float  v8f;

union ABFrag { v16bf v; unsigned int u[8]; };

// ---------------- async global->LDS staging (CDNA5 path with fallback) ------
#if defined(__has_builtin)
#if __has_builtin(__builtin_amdgcn_global_load_async_to_lds_b128)
#define USE_ASYNC_LDS 1
#endif
#endif

#ifdef USE_ASYNC_LDS
typedef int vi4 __attribute__((vector_size(16)));
typedef __attribute__((address_space(1))) vi4* gvi4_t;
typedef __attribute__((address_space(3))) vi4* lvi4_t;
#define ASYNC_B128(g, l, off) \
  __builtin_amdgcn_global_load_async_to_lds_b128((gvi4_t)(g), (lvi4_t)(l), (off), 0)
#endif

__device__ __forceinline__ void stage16B(const unsigned short* gsrc, unsigned short* ldst) {
#ifdef USE_ASYNC_LDS
  ASYNC_B128(gsrc, ldst, 0);
#else
  *(uint4*)ldst = *(const uint4*)gsrc;
#endif
}
__device__ __forceinline__ void stage32B(const unsigned short* gsrc, unsigned short* ldst) {
#ifdef USE_ASYNC_LDS
  ASYNC_B128(gsrc, ldst, 0);
  ASYNC_B128(gsrc, ldst, 16);
#else
  const uint4* s = (const uint4*)gsrc; uint4* d = (uint4*)ldst;
  d[0] = s[0]; d[1] = s[1];
#endif
}
__device__ __forceinline__ void stage64B(const unsigned short* gsrc, unsigned short* ldst) {
#ifdef USE_ASYNC_LDS
  ASYNC_B128(gsrc, ldst, 0);
  ASYNC_B128(gsrc, ldst, 16);
  ASYNC_B128(gsrc, ldst, 32);
  ASYNC_B128(gsrc, ldst, 48);
#else
  const uint4* s = (const uint4*)gsrc; uint4* d = (uint4*)ldst;
  d[0] = s[0]; d[1] = s[1]; d[2] = s[2]; d[3] = s[3];
#endif
}

// Wait until at most N of this wave's async ops remain in flight.  Async loads
// complete in order, so ASYNCcnt<=N proves the older (current-buffer) tile landed
// while the newer (next-buffer) prefetch keeps flying.
template <int N>
__device__ __forceinline__ void async_join_n() {
#ifdef USE_ASYNC_LDS
#if __has_builtin(__builtin_amdgcn_s_wait_asynccnt)
  __builtin_amdgcn_s_wait_asynccnt(N);
#else
  asm volatile("s_wait_asynccnt %0" ::"i"(N) : "memory");
#endif
#endif
}

// ---------------------------------------------------------------------------
__device__ __forceinline__ unsigned short f2bf(float x) {
  unsigned int t = __float_as_uint(x);
  t += 0x7fffu + ((t >> 16) & 1u);           // round-to-nearest-even
  return (unsigned short)(t >> 16);
}

// A fragment (16x32 bf16) from row-major src[M][K].  lane: m = caller-passed row,
// half h = lane>>4.  element e: K = e + 8h (e<8), K = e + 8 + 8h (e>=8).
__device__ __forceinline__ void load_a_frag(ABFrag& f, const unsigned short* src,
                                            int m, int ld, int lane) {
  const unsigned short* p = src + m * ld + ((lane >> 4) << 3);
#pragma unroll
  for (int j = 0; j < 4; ++j) f.u[j] = *(const unsigned int*)(p + 2 * j);
#pragma unroll
  for (int j = 4; j < 8; ++j) f.u[j] = *(const unsigned int*)(p + 2 * j + 8);
}

// B fragment (32x16 bf16) from row-major src[COL][K].  lane: col, K = e + 8h (e<8), K = e + 8 + 8h (e>=8).
__device__ __forceinline__ void load_b_frag(ABFrag& f, const unsigned short* src,
                                            int col, int ld, int lane) {
  const unsigned short* p = src + col * ld + ((lane >> 4) << 3);
#pragma unroll
  for (int j = 0; j < 4; ++j) f.u[j] = *(const unsigned int*)(p + 2 * j);
#pragma unroll
  for (int j = 4; j < 8; ++j) f.u[j] = *(const unsigned int*)(p + 2 * j + 8);
}

__device__ __forceinline__ v8f wmma_bf(const ABFrag& a, const ABFrag& b, v8f c) {
  return __builtin_amdgcn_wmma_f32_16x16x32_bf16(false, a.v, false, b.v,
                                                 (short)0, c, false, false);
}

// ---------------------------------------------------------------- kernel 1
// Pack wq|wk|wv into one bf16 [3C,C] A-matrix, and wp into bf16 [C,C].
__global__ __launch_bounds__(256) void k_wconv(
    const float* wq, const float* wk, const float* wv, const float* wp,
    unsigned short* wqkv, unsigned short* wpb) {
  int i = blockIdx.x * 256 + threadIdx.x;          // 4*CH*CH total
  const int cc = CH * CH;
  if (i < cc)            wqkv[i]        = f2bf(wq[i]);
  else if (i < 2 * cc)   wqkv[i]        = f2bf(wk[i - cc]);
  else if (i < 3 * cc)   wqkv[i]        = f2bf(wv[i - 2 * cc]);
  else                   wpb[i - 3*cc]  = f2bf(wp[i - 3 * cc]);
}

// ---------------------------------------------------------------- kernel 2
// GroupNorm over (8 ch x 1024 px) contiguous groups; write hT[b][n][c] bf16.
__global__ __launch_bounds__(256) void k_gnorm(
    const float* x, const float* gamma, const float* beta, unsigned short* hT) {
  int b = blockIdx.x >> 5, g = blockIdx.x & 31;
  const float* xp = x + ((size_t)b * CH + g * CPG) * NP;   // 8192 contiguous
  float s = 0.f, s2 = 0.f;
  for (int i = threadIdx.x; i < CPG * NP; i += 256) {
    float v = xp[i]; s += v; s2 += v * v;
  }
  __shared__ float red0[256], red1[256];
  red0[threadIdx.x] = s; red1[threadIdx.x] = s2;
  __syncthreads();
  for (int st = 128; st > 0; st >>= 1) {
    if (threadIdx.x < st) {
      red0[threadIdx.x] += red0[threadIdx.x + st];
      red1[threadIdx.x] += red1[threadIdx.x + st];
    }
    __syncthreads();
  }
  float mean = red0[0] * (1.0f / 8192.0f);
  float var  = red1[0] * (1.0f / 8192.0f) - mean * mean;
  float rstd = rsqrtf(var + 1e-5f);
  for (int i = threadIdx.x; i < CPG * NP; i += 256) {
    int c = g * CPG + (i >> 10), n = i & 1023;
    float hv = (xp[i] - mean) * rstd * gamma[c] + beta[c];
    hT[((size_t)b * NP + n) * CH + c] = f2bf(hv);
  }
}

// ---------------------------------------------------------------- kernel 3
// Fused QKV GEMM: [768,256] x [256,1024].  Block tile 64x128, 8 waves of 32x32.
// Double-buffered async LDS staging (3 async ops per wave per stage).
__global__ __launch_bounds__(256) void k_qkv(
    const unsigned short* wqkv, const unsigned short* hT,
    const float* bq, const float* bk, const float* bv,
    unsigned short* qT, unsigned short* kT, unsigned short* vout) {
  int mt = blockIdx.x, nt = blockIdx.y, b = blockIdx.z;
  int tid = threadIdx.x, lane = tid & 31, w = tid >> 5;
  __shared__ alignas(16) unsigned short Al[2][64 * 32];
  __shared__ alignas(16) unsigned short Bl[2][128 * 32];
  int m_w = (w >> 2) * 32, n_w = (w & 3) * 32;
  v8f acc[2][2] = {};
  const unsigned short* hB = hT + (size_t)b * NP * CH;

  auto stage = [&](int buf, int kk2) {
    int ra = tid >> 2, ca = (tid & 3) * 8;
    stage16B(&wqkv[(size_t)(mt * 64 + ra) * CH + kk2 + ca], &Al[buf][ra * 32 + ca]);
    int rb = tid >> 1, cb = (tid & 1) * 16;
    stage32B(hB + (size_t)(nt * 128 + rb) * CH + kk2 + cb, &Bl[buf][rb * 32 + cb]);
  };

  stage(0, 0);
#pragma unroll
  for (int kk = 0; kk < CH; kk += 32) {
    int cur = (kk >> 5) & 1;
    if (kk + 32 < CH) { stage(1 - cur, kk + 32); async_join_n<3>(); }
    else              { async_join_n<0>(); }
    __syncthreads();
    ABFrag a0, a1, b0, b1;
    load_a_frag(a0, Al[cur], m_w + (lane & 15), 32, lane);
    load_a_frag(a1, Al[cur], m_w + 16 + (lane & 15), 32, lane);
    load_b_frag(b0, Bl[cur], n_w + (lane & 15), 32, lane);
    load_b_frag(b1, Bl[cur], n_w + 16 + (lane & 15), 32, lane);
    acc[0][0] = wmma_bf(a0, b0, acc[0][0]);
    acc[0][1] = wmma_bf(a0, b1, acc[0][1]);
    acc[1][0] = wmma_bf(a1, b0, acc[1][0]);
    acc[1][1] = wmma_bf(a1, b1, acc[1][1]);
    __syncthreads();
  }

  int h4 = lane >> 4;
#pragma unroll
  for (int fm = 0; fm < 2; ++fm)
#pragma unroll
    for (int fn = 0; fn < 2; ++fn) {
      int gcol = nt * 128 + n_w + fn * 16 + (lane & 15);
#pragma unroll
      for (int r = 0; r < 8; ++r) {
        int grow = mt * 64 + m_w + fm * 16 + r + 8 * h4;
        int cidx = grow & 255;
        float val = acc[fm][fn][r];
        if (grow < 256) {
          val = (val + bq[cidx]) * 0.0625f;               // fold 1/sqrt(256)
          qT[((size_t)b * NP + gcol) * CH + cidx] = f2bf(val);
        } else if (grow < 512) {
          val += bk[cidx];
          kT[((size_t)b * NP + gcol) * CH + cidx] = f2bf(val);
        } else {
          val += bv[cidx];
          vout[((size_t)b * CH + cidx) * NP + gcol] = f2bf(val);
        }
      }
    }
}

// ---------------------------------------------------------------- kernel 4
// Flash attention: block = 128 query rows (8 waves x 16 rows), KV tiles of 32,
// double-buffered async staging (8 async ops per wave per stage).
__global__ __launch_bounds__(256) void k_attn(
    const unsigned short* qT, const unsigned short* kT,
    const unsigned short* vmat, unsigned short* oT) {
  int b = blockIdx.y, m0 = blockIdx.x * 128;
  int tid = threadIdx.x, lane = tid & 31, w = tid >> 5;
  __shared__ alignas(16) unsigned short Kl[2][32 * 256];   // [n][c]
  __shared__ alignas(16) unsigned short Vl[2][256 * 32];   // [c][n]
  __shared__ alignas(16) unsigned short Pl[8][16 * 32];    // per-wave P [m][n]
  const unsigned short* qB = qT   + (size_t)b * NP * CH;
  const unsigned short* kB = kT   + (size_t)b * NP * CH;
  const unsigned short* vB = vmat + (size_t)b * CH * NP;

  auto stageKV = [&](int buf, int n1) {
    int r = tid >> 3, c = (tid & 7) * 32;
    stage64B(kB + (size_t)(n1 + r) * CH + c, &Kl[buf][r * 256 + c]);
    stage64B(vB + (size_t)tid * NP + n1, &Vl[buf][tid * 32]);
  };

  // Q fragments resident in VGPRs (16 rows x 256 channels = 8 frags)
  ABFrag qf[8];
  int mrow = m0 + w * 16 + (lane & 15);
#pragma unroll
  for (int c8 = 0; c8 < 8; ++c8) load_a_frag(qf[c8], qB + c8 * 32, mrow, CH, lane);

  v8f o[16] = {};
  float rm[8], rl[8];
#pragma unroll
  for (int r = 0; r < 8; ++r) { rm[r] = -1e30f; rl[r] = 0.f; }

  stageKV(0, 0);
  for (int n0 = 0; n0 < NP; n0 += 32) {
    int cur = (n0 >> 5) & 1;
    if (n0 + 32 < NP) { stageKV(1 - cur, n0 + 32); async_join_n<8>(); }
    else              { async_join_n<0>(); }
    __syncthreads();

    // scores: two 16x16 tiles over K=256
    v8f s0 = {}, s1 = {};
#pragma unroll
    for (int c8 = 0; c8 < 8; ++c8) {
      ABFrag b0, b1;
      load_b_frag(b0, Kl[cur] + c8 * 32, (lane & 15), 256, lane);
      load_b_frag(b1, Kl[cur] + c8 * 32, 16 + (lane & 15), 256, lane);
      s0 = wmma_bf(qf[c8], b0, s0);
      s1 = wmma_bf(qf[c8], b1, s1);
    }

    // online softmax; row reduce across 16-lane halves via shfl_xor
    unsigned short pbits[16];
    float corr[8];
#pragma unroll
    for (int r = 0; r < 8; ++r) {
      float mx = fmaxf(s0[r], s1[r]);
      for (int off = 8; off > 0; off >>= 1) mx = fmaxf(mx, __shfl_xor(mx, off));
      float mnew = fmaxf(rm[r], mx);
      float p0 = __expf(s0[r] - mnew);
      float p1 = __expf(s1[r] - mnew);
      float ps = p0 + p1;
      for (int off = 8; off > 0; off >>= 1) ps += __shfl_xor(ps, off);
      float cf = __expf(rm[r] - mnew);
      rl[r] = rl[r] * cf + ps;
      rm[r] = mnew;
      corr[r] = cf;
      pbits[r] = f2bf(p0); pbits[8 + r] = f2bf(p1);
    }
#pragma unroll
    for (int ct = 0; ct < 16; ++ct)
#pragma unroll
      for (int r = 0; r < 8; ++r) o[ct][r] *= corr[r];

    { // transpose P (C-layout -> A-layout) through per-wave LDS
      unsigned short* pw = &Pl[w][0];
      int hh = lane >> 4, nc = lane & 15;
#pragma unroll
      for (int r = 0; r < 8; ++r) {
        pw[(r + 8 * hh) * 32 + nc]      = pbits[r];
        pw[(r + 8 * hh) * 32 + 16 + nc] = pbits[8 + r];
      }
    }
    __syncthreads();
    ABFrag pf;
    load_a_frag(pf, &Pl[w][0], (lane & 15), 32, lane);
#pragma unroll
    for (int ct = 0; ct < 16; ++ct) {
      ABFrag bvf;
      load_b_frag(bvf, Vl[cur] + ct * 16 * 32, (lane & 15), 32, lane);
      o[ct] = wmma_bf(pf, bvf, o[ct]);
    }
    __syncthreads();
  }

  // normalize and write oT[b][m][c]
  unsigned short* oB = oT + (size_t)b * NP * CH;
  int hh = lane >> 4, nc = lane & 15;
#pragma unroll
  for (int r = 0; r < 8; ++r) {
    float inv = 1.0f / rl[r];
    int m = m0 + w * 16 + r + 8 * hh;
#pragma unroll
    for (int ct = 0; ct < 16; ++ct)
      oB[(size_t)m * CH + ct * 16 + nc] = f2bf(o[ct][r] * inv);
  }
}

// ---------------------------------------------------------------- kernel 5
// Projection [256,256] x [256,1024] + bias + residual, f32 out.
__global__ __launch_bounds__(256) void k_proj(
    const unsigned short* wpb, const unsigned short* oT,
    const float* bp, const float* x, float* out) {
  int mt = blockIdx.x, nt = blockIdx.y, b = blockIdx.z;
  int tid = threadIdx.x, lane = tid & 31, w = tid >> 5;
  __shared__ alignas(16) unsigned short Al[2][64 * 32];
  __shared__ alignas(16) unsigned short Bl[2][128 * 32];
  int m_w = (w >> 2) * 32, n_w = (w & 3) * 32;
  v8f acc[2][2] = {};
  const unsigned short* oB = oT + (size_t)b * NP * CH;

  auto stage = [&](int buf, int kk2) {
    int ra = tid >> 2, ca = (tid & 3) * 8;
    stage16B(&wpb[(size_t)(mt * 64 + ra) * CH + kk2 + ca], &Al[buf][ra * 32 + ca]);
    int rb = tid >> 1, cb = (tid & 1) * 16;
    stage32B(oB + (size_t)(nt * 128 + rb) * CH + kk2 + cb, &Bl[buf][rb * 32 + cb]);
  };

  stage(0, 0);
#pragma unroll
  for (int kk = 0; kk < CH; kk += 32) {
    int cur = (kk >> 5) & 1;
    if (kk + 32 < CH) { stage(1 - cur, kk + 32); async_join_n<3>(); }
    else              { async_join_n<0>(); }
    __syncthreads();
    ABFrag a0, a1, b0, b1;
    load_a_frag(a0, Al[cur], m_w + (lane & 15), 32, lane);
    load_a_frag(a1, Al[cur], m_w + 16 + (lane & 15), 32, lane);
    load_b_frag(b0, Bl[cur], n_w + (lane & 15), 32, lane);
    load_b_frag(b1, Bl[cur], n_w + 16 + (lane & 15), 32, lane);
    acc[0][0] = wmma_bf(a0, b0, acc[0][0]);
    acc[0][1] = wmma_bf(a0, b1, acc[0][1]);
    acc[1][0] = wmma_bf(a1, b0, acc[1][0]);
    acc[1][1] = wmma_bf(a1, b1, acc[1][1]);
    __syncthreads();
  }

  int h4 = lane >> 4;
#pragma unroll
  for (int fm = 0; fm < 2; ++fm)
#pragma unroll
    for (int fn = 0; fn < 2; ++fn) {
      int gcol = nt * 128 + n_w + fn * 16 + (lane & 15);
#pragma unroll
      for (int r = 0; r < 8; ++r) {
        int grow = mt * 64 + m_w + fm * 16 + r + 8 * h4;
        size_t idx = ((size_t)b * CH + grow) * NP + gcol;
        out[idx] = x[idx] + acc[fm][fn][r] + bp[grow];
      }
    }
}

// ---------------------------------------------------------------- launch
extern "C" void kernel_launch(void* const* d_in, const int* in_sizes, int n_in,
                              void* d_out, int out_size, void* d_ws, size_t ws_size,
                              hipStream_t stream) {
  const float* x     = (const float*)d_in[0];
  const float* gamma = (const float*)d_in[1];
  const float* beta  = (const float*)d_in[2];
  const float* wq    = (const float*)d_in[3];
  const float* bq    = (const float*)d_in[4];
  const float* wk    = (const float*)d_in[5];
  const float* bk    = (const float*)d_in[6];
  const float* wv    = (const float*)d_in[7];
  const float* bv    = (const float*)d_in[8];
  const float* wp    = (const float*)d_in[9];
  const float* bp    = (const float*)d_in[10];
  float* out = (float*)d_out;

  char* ws = (char*)d_ws;
  size_t off = 0;
  auto take = [&](size_t bytes) {
    char* p = ws + off;
    off += (bytes + 255) & ~(size_t)255;
    return p;
  };
  unsigned short* wqkv = (unsigned short*)take((size_t)3 * CH * CH * 2);
  unsigned short* wpb  = (unsigned short*)take((size_t)CH * CH * 2);
  unsigned short* hT   = (unsigned short*)take((size_t)NB * NP * CH * 2);
  unsigned short* qTb  = (unsigned short*)take((size_t)NB * NP * CH * 2);
  unsigned short* kTb  = (unsigned short*)take((size_t)NB * NP * CH * 2);
  unsigned short* vb   = (unsigned short*)take((size_t)NB * CH * NP * 2);
  unsigned short* oTb  = (unsigned short*)take((size_t)NB * NP * CH * 2);

  k_wconv<<<(4 * CH * CH) / 256, 256, 0, stream>>>(wq, wk, wv, wp, wqkv, wpb);
  k_gnorm<<<NB * NG, 256, 0, stream>>>(x, gamma, beta, hT);
  k_qkv<<<dim3(12, 8, NB), 256, 0, stream>>>(wqkv, hT, bq, bk, bv, qTb, kTb, vb);
  k_attn<<<dim3(8, NB), 256, 0, stream>>>(qTb, kTb, vb, oTb);
  k_proj<<<dim3(4, 8, NB), 256, 0, stream>>>(wpb, oTb, bp, x, out);
}
